// CausalSelfAttention_26525718020693
// MI455X (gfx1250) — matrix operand order repaired and checked by reading, not compiled, not run on hardware
//
#include <hip/hip_runtime.h>
#include <hip/hip_bf16.h>

// ---------------- problem constants ----------------
constexpr int CM    = 1024;           // d_model
constexpr int SEQ   = 2048;           // T
constexpr int BATCH = 4;
constexpr int NH    = 16;
constexpr int DH    = 64;
constexpr int MTOT  = BATCH * SEQ;    // 8192 rows
constexpr int N_QKV = 3 * CM;         // 3072

// ---------------- WMMA types / helpers ----------------
typedef __attribute__((ext_vector_type(16))) __bf16 v16bf;
typedef __attribute__((ext_vector_type(8)))  float  v8f;
typedef __attribute__((ext_vector_type(4)))  int    v4i;

#define AS1 __attribute__((address_space(1)))
#define AS3 __attribute__((address_space(3)))

#if __has_builtin(__builtin_amdgcn_global_load_async_to_lds_b128)
#define HAVE_ASYNC 1
#endif

struct FragU { uint4 lo, hi; };

__device__ __forceinline__ v16bf frag_cast(FragU f) {
    return __builtin_bit_cast(v16bf, f);
}

__device__ __forceinline__ v8f zero8() {
    v8f z;
#pragma unroll
    for (int i = 0; i < 8; ++i) z[i] = 0.0f;
    return z;
}

__device__ __forceinline__ v8f wmma_bf16(v16bf a, v16bf b, v8f c) {
    // D = A(16x32 bf16) x B(32x16 bf16) + C(16x16 f32)
    return __builtin_amdgcn_wmma_f32_16x16x32_bf16(
        /*neg_a=*/false, a, /*neg_b=*/false, b,
        /*c_mod=*/(short)0, c, /*reuse_a=*/false, /*reuse_b=*/false);
}

// float -> bf16 bits, round-to-nearest-even
__device__ __forceinline__ unsigned short f2bf(float x) {
    unsigned int u = __float_as_uint(x);
    u += 0x7fffu + ((u >> 16) & 1u);
    return (unsigned short)(u >> 16);
}

// 16B-per-lane async copy global->LDS (ASYNCcnt path on gfx1250)
__device__ __forceinline__ void async_b128(const unsigned short* g, unsigned short* l) {
#ifdef HAVE_ASYNC
    __builtin_amdgcn_global_load_async_to_lds_b128(
        (AS1 v4i*)const_cast<unsigned short*>(g), (AS3 v4i*)l, 0, 0);
#else
    *(uint4*)l = *(const uint4*)g;
#endif
}

__device__ __forceinline__ void wait_async0() {
#ifdef HAVE_ASYNC
#if __has_builtin(__builtin_amdgcn_s_wait_asynccnt)
    __builtin_amdgcn_s_wait_asynccnt(0);
#else
    asm volatile("s_wait_asynccnt 0" ::: "memory");
#endif
#endif
}

// A fragment (16-bit 16x32, ISA 7.12.2): m = lane%16, grp = lane/16.
// element e -> K = kbase + e + (e>=8 ? 8 : 0) + 8*grp  => two contiguous 8-half runs.
__device__ __forceinline__ v16bf lds_frag_a(const unsigned short* base, int row,
                                            int stride, int kbase) {
    int lane = threadIdx.x & 31;
    int m = lane & 15, grp = lane >> 4;
    const unsigned short* p = base + (size_t)(row + m) * stride + kbase + grp * 8;
    FragU f;
    f.lo = *(const uint4*)(p);
    f.hi = *(const uint4*)(p + 16);
    return frag_cast(f);
}

// B fragment (16-bit 32x16): element e -> K = kbase + e + (e>=8 ? 8 : 0) + 8*grp, N = ncol + lane%16.
// LDS laid out [N][K] with stride strideK -> two contiguous 8-half runs.
__device__ __forceinline__ v16bf lds_frag_b(const unsigned short* base, int ncol,
                                            int strideK, int kbase) {
    int lane = threadIdx.x & 31;
    int n = lane & 15, grp = lane >> 4;
    const unsigned short* p = base + (size_t)(ncol + n) * strideK + kbase + grp * 8;
    FragU f;
    f.lo = *(const uint4*)(p);
    f.hi = *(const uint4*)(p + 16);
    return frag_cast(f);
}

// ==========================================================================
// Kernel 1: qkv = x @ w_attn + b_attn ; scatter q/k/v to [B,H,T,Dh] bf16
//   128x128 block tile, BK=32, double-buffered LDS, 8 waves x (32x64)
// ==========================================================================
__global__ __launch_bounds__(256) void qkv_gemm(
    const float* __restrict__ x, const float* __restrict__ w,
    const float* __restrict__ bias,
    unsigned short* __restrict__ qb, unsigned short* __restrict__ kb,
    unsigned short* __restrict__ vb)
{
    __shared__ __align__(16) unsigned short xs[2][128 * 32];  // [m][k]
    __shared__ __align__(16) unsigned short wt[2][128 * 32];  // [n][k] (transposed)

    const int tid = threadIdx.x;
    const int wave = tid >> 5, lane = tid & 31;
    const int m0 = blockIdx.x * 128;
    const int n0 = blockIdx.y * 128;
    const int wr = (wave & 3) * 32;
    const int wc = (wave >> 2) * 64;

    auto stage = [&](int k0, int buf) {
        // X tile 128x32 (fp32 -> bf16)
#pragma unroll
        for (int it = 0; it < 4; ++it) {
            int j  = tid + 256 * it;
            int r  = j >> 3;
            int c4 = (j & 7) << 2;
            float4 v = *(const float4*)(x + (size_t)(m0 + r) * CM + k0 + c4);
            unsigned short* d = &xs[buf][r * 32 + c4];
            d[0] = f2bf(v.x); d[1] = f2bf(v.y); d[2] = f2bf(v.z); d[3] = f2bf(v.w);
        }
        // W tile 32x128 transposed -> wt[n][k]
#pragma unroll
        for (int it = 0; it < 4; ++it) {
            int j  = tid + 256 * it;
            int r  = j >> 5;
            int c4 = (j & 31) << 2;
            float4 v = *(const float4*)(w + (size_t)(k0 + r) * N_QKV + n0 + c4);
            wt[buf][(c4 + 0) * 32 + r] = f2bf(v.x);
            wt[buf][(c4 + 1) * 32 + r] = f2bf(v.y);
            wt[buf][(c4 + 2) * 32 + r] = f2bf(v.z);
            wt[buf][(c4 + 3) * 32 + r] = f2bf(v.w);
        }
    };

    v8f acc[2][4];
#pragma unroll
    for (int i = 0; i < 2; ++i)
#pragma unroll
        for (int j = 0; j < 4; ++j) acc[i][j] = zero8();

    stage(0, 0);
    __syncthreads();

    int buf = 0;
    for (int k0 = 0; k0 < CM; k0 += 32) {
        if (k0 + 64 < CM) {  // prefetch tile two steps ahead
            __builtin_prefetch(x + (size_t)(m0 + (tid >> 1)) * CM + k0 + 64 + (tid & 1) * 16, 0, 0);
            __builtin_prefetch(w + (size_t)(k0 + 64 + (tid >> 3)) * N_QKV + n0 + (tid & 7) * 16, 0, 0);
        }
        if (k0 + 32 < CM) stage(k0 + 32, buf ^ 1);

        v16bf a[2], b[4];
#pragma unroll
        for (int i = 0; i < 2; ++i) a[i] = lds_frag_a(xs[buf], wr + 16 * i, 32, 0);
#pragma unroll
        for (int j = 0; j < 4; ++j) b[j] = lds_frag_b(wt[buf], wc + 16 * j, 32, 0);
#pragma unroll
        for (int i = 0; i < 2; ++i)
#pragma unroll
            for (int j = 0; j < 4; ++j)
                acc[i][j] = wmma_bf16(a[i], b[j], acc[i][j]);
        __syncthreads();
        buf ^= 1;
    }

    // epilogue: + bias, scatter into q/k/v [B,H,T,Dh] bf16
    const int mlane = lane & 15, grp = lane >> 4;
#pragma unroll
    for (int i = 0; i < 2; ++i) {
#pragma unroll
        for (int j = 0; j < 4; ++j) {
            int n   = n0 + wc + 16 * j + mlane;
            float bv = bias[n];
            int sel = n >> 10;               // 0:q 1:k 2:v
            int h   = (n & 1023) >> 6;
            int dh  = n & 63;
            unsigned short* dst = (sel == 0) ? qb : ((sel == 1) ? kb : vb);
#pragma unroll
            for (int r = 0; r < 8; ++r) {
                int m = m0 + wr + 16 * i + r + 8 * grp;
                int bidx = m >> 11;
                int t    = m & 2047;
                dst[(((size_t)(bidx * NH + h)) * SEQ + t) * DH + dh] =
                    f2bf(acc[i][j][r] + bv);
            }
        }
    }
}

// ==========================================================================
// Kernel 2: flash attention per (b,h); y written as [B,T,C] bf16
//   Async K staging + double-buffered K/V blocks of 32 keys
// ==========================================================================
__global__ __launch_bounds__(256) void attn_kernel(
    const unsigned short* __restrict__ qb, const unsigned short* __restrict__ kb,
    const unsigned short* __restrict__ vb, unsigned short* __restrict__ yb)
{
    __shared__ __align__(16) unsigned short qs[128 * 64];      // [t][dh]
    __shared__ __align__(16) unsigned short ks[2][32 * 64];    // [key][dh]
    __shared__ __align__(16) unsigned short vt[2][64 * 32];    // [dh][key]
    __shared__ __align__(16) unsigned short ps[8][16 * 32];    // per-wave P scratch

    const int tid = threadIdx.x, wave = tid >> 5, lane = tid & 31;
    const int mlane = lane & 15, grp = lane >> 4;
    const int bh = blockIdx.y;
    const int t0 = blockIdx.x * 128;
    const size_t base = (size_t)bh * SEQ * DH;

    auto stage_k = [&](int kb0, int buf) {
        async_b128(kb + base + (size_t)kb0 * DH + (size_t)tid * 8, &ks[buf][tid * 8]);
    };
    auto stage_v = [&](int kb0, int buf) {   // transpose -> vt[dh][key]
        int key = tid >> 3;
        int d0  = (tid & 7) * 8;
        uint4 v4 = *(const uint4*)(vb + base + (size_t)(kb0 + key) * DH + d0);
        const unsigned short* hp = (const unsigned short*)&v4;
#pragma unroll
        for (int e = 0; e < 8; ++e) vt[buf][(d0 + e) * 32 + key] = hp[e];
    };

    // stage Q block 128x64 (async) + first K/V block
#pragma unroll
    for (int it = 0; it < 4; ++it) {
        int j = tid + 256 * it;
        async_b128(qb + base + (size_t)t0 * DH + (size_t)j * 8, &qs[j * 8]);
    }
    stage_k(0, 0);
    stage_v(0, 0);
    wait_async0();
    __syncthreads();

    v16bf qa[2];
    qa[0] = lds_frag_a(qs, wave * 16, 64, 0);    // dh 0..31
    qa[1] = lds_frag_a(qs, wave * 16, 64, 32);   // dh 32..63

    float mrow[8], lrow[8];
#pragma unroll
    for (int r = 0; r < 8; ++r) { mrow[r] = -1e30f; lrow[r] = 0.0f; }
    v8f acc[4];
#pragma unroll
    for (int j = 0; j < 4; ++j) acc[j] = zero8();

    int buf = 0;
    for (int kb0 = 0; kb0 < SEQ; kb0 += 32) {
        if (kb0 + 32 < SEQ) {               // stage next block into other buffer
            stage_k(kb0 + 32, buf ^ 1);
            stage_v(kb0 + 32, buf ^ 1);
        }

        // S = (Q K^T) * scale for two 16-key subtiles
        v8f s[2];
#pragma unroll
        for (int sub = 0; sub < 2; ++sub) {
            v16bf b0 = lds_frag_b(ks[buf], sub * 16, 64, 0);
            v16bf b1 = lds_frag_b(ks[buf], sub * 16, 64, 32);
            v8f c = zero8();
            c = wmma_bf16(qa[0], b0, c);
            c = wmma_bf16(qa[1], b1, c);
            s[sub] = c;
        }

        // online softmax: rows r+8*grp, cols spread over 16 lanes of the group
        const float scale = 0.125f;          // 1/sqrt(64)
        float corr[8];
#pragma unroll
        for (int r = 0; r < 8; ++r) {
            float v0 = s[0][r] * scale, v1 = s[1][r] * scale;
            float mx = fmaxf(v0, v1);
#pragma unroll
            for (int off = 1; off < 16; off <<= 1)
                mx = fmaxf(mx, __shfl_xor(mx, off, 32));
            float mnew = fmaxf(mrow[r], mx);
            corr[r] = __expf(mrow[r] - mnew);
            mrow[r] = mnew;
            float p0 = __expf(v0 - mnew), p1 = __expf(v1 - mnew);
            s[0][r] = p0; s[1][r] = p1;
            float rs = p0 + p1;
#pragma unroll
            for (int off = 1; off < 16; off <<= 1)
                rs += __shfl_xor(rs, off, 32);
            lrow[r] = lrow[r] * corr[r] + rs;
        }

        // P -> per-wave LDS scratch (A-layout source), rescale accumulators
        unsigned short* pw = ps[wave];
#pragma unroll
        for (int r = 0; r < 8; ++r) {
            int row = r + 8 * grp;
            pw[row * 32 + mlane]      = f2bf(s[0][r]);
            pw[row * 32 + 16 + mlane] = f2bf(s[1][r]);
            float cr = corr[r];
#pragma unroll
            for (int j = 0; j < 4; ++j) acc[j][r] *= cr;
        }
        asm volatile("s_wait_dscnt 0" ::: "memory");  // same-wave LDS RAW

        v16bf pa = lds_frag_a(pw, 0, 32, 0);          // P 16x32
#pragma unroll
        for (int j = 0; j < 4; ++j) {
            v16bf bv = lds_frag_b(vt[buf], j * 16, 32, 0);
            acc[j] = wmma_bf16(pa, bv, acc[j]);
        }

        wait_async0();          // next K block resident in LDS
        __syncthreads();        // all waves done with current buffers
        buf ^= 1;
    }

    // y[b][t][h*64+dh] = acc / l
    const int b = bh >> 4, h = bh & 15;
#pragma unroll
    for (int j = 0; j < 4; ++j) {
#pragma unroll
        for (int r = 0; r < 8; ++r) {
            int t = t0 + wave * 16 + r + 8 * grp;
            float val = acc[j][r] / lrow[r];
            yb[((size_t)(b * SEQ + t)) * CM + h * DH + j * 16 + mlane] = f2bf(val);
        }
    }
}

// ==========================================================================
// Kernel 3: out = y @ w_proj + b_proj (fp32 out), async y staging
// ==========================================================================
__global__ __launch_bounds__(256) void proj_gemm(
    const unsigned short* __restrict__ y, const float* __restrict__ w,
    const float* __restrict__ bias, float* __restrict__ out)
{
    __shared__ __align__(16) unsigned short ys[2][128 * 32];
    __shared__ __align__(16) unsigned short wt[2][128 * 32];

    const int tid = threadIdx.x, wave = tid >> 5, lane = tid & 31;
    const int m0 = blockIdx.x * 128, n0 = blockIdx.y * 128;
    const int wr = (wave & 3) * 32, wc = (wave >> 2) * 64;

    auto stage = [&](int k0, int buf) {
        // y tile 128x32 bf16: b128 async copies
#pragma unroll
        for (int it = 0; it < 2; ++it) {
            int j  = tid + 256 * it;
            int r  = j >> 2;
            int c8 = (j & 3) * 8;
            async_b128(y + (size_t)(m0 + r) * CM + k0 + c8, &ys[buf][r * 32 + c8]);
        }
        // w_proj tile 32x128 transposed (fp32 -> bf16)
#pragma unroll
        for (int it = 0; it < 4; ++it) {
            int j  = tid + 256 * it;
            int r  = j >> 5;
            int c4 = (j & 31) << 2;
            float4 v = *(const float4*)(w + (size_t)(k0 + r) * CM + n0 + c4);
            wt[buf][(c4 + 0) * 32 + r] = f2bf(v.x);
            wt[buf][(c4 + 1) * 32 + r] = f2bf(v.y);
            wt[buf][(c4 + 2) * 32 + r] = f2bf(v.z);
            wt[buf][(c4 + 3) * 32 + r] = f2bf(v.w);
        }
    };

    v8f acc[2][4];
#pragma unroll
    for (int i = 0; i < 2; ++i)
#pragma unroll
        for (int j = 0; j < 4; ++j) acc[i][j] = zero8();

    stage(0, 0);
    wait_async0();
    __syncthreads();

    int buf = 0;
    for (int k0 = 0; k0 < CM; k0 += 32) {
        if (k0 + 64 < CM)
            __builtin_prefetch(w + (size_t)(k0 + 64 + (tid >> 3)) * CM + n0 + (tid & 7) * 16, 0, 0);
        if (k0 + 32 < CM) stage(k0 + 32, buf ^ 1);

        v16bf a[2], b[4];
#pragma unroll
        for (int i = 0; i < 2; ++i) a[i] = lds_frag_a(ys[buf], wr + 16 * i, 32, 0);
#pragma unroll
        for (int j = 0; j < 4; ++j) b[j] = lds_frag_b(wt[buf], wc + 16 * j, 32, 0);
#pragma unroll
        for (int i = 0; i < 2; ++i)
#pragma unroll
            for (int j = 0; j < 4; ++j)
                acc[i][j] = wmma_bf16(a[i], b[j], acc[i][j]);

        wait_async0();
        __syncthreads();
        buf ^= 1;
    }

    const int mlane = lane & 15, grp = lane >> 4;
#pragma unroll
    for (int i = 0; i < 2; ++i) {
#pragma unroll
        for (int j = 0; j < 4; ++j) {
            int n = n0 + wc + 16 * j + mlane;
            float bv = bias[n];
#pragma unroll
            for (int r = 0; r < 8; ++r) {
                int m = m0 + wr + 16 * i + r + 8 * grp;
                out[(size_t)m * CM + n] = acc[i][j][r] + bv;
            }
        }
    }
}

// ==========================================================================
extern "C" void kernel_launch(void* const* d_in, const int* in_sizes, int n_in,
                              void* d_out, int out_size, void* d_ws, size_t ws_size,
                              hipStream_t stream) {
    const float* x      = (const float*)d_in[0];
    const float* w_attn = (const float*)d_in[1];
    const float* b_attn = (const float*)d_in[2];
    const float* w_proj = (const float*)d_in[3];
    const float* b_proj = (const float*)d_in[4];
    float* out = (float*)d_out;

    const size_t elems = (size_t)MTOT * CM;
    unsigned short* ws = (unsigned short*)d_ws;
    unsigned short* qb = ws;
    unsigned short* kb = ws + elems;
    unsigned short* vb = ws + 2 * elems;
    unsigned short* yb = ws + 3 * elems;

    dim3 blk(256);
    qkv_gemm<<<dim3(MTOT / 128, N_QKV / 128), blk, 0, stream>>>(
        x, w_attn, b_attn, qb, kb, vb);
    attn_kernel<<<dim3(SEQ / 128, BATCH * NH), blk, 0, stream>>>(
        qb, kb, vb, yb);
    proj_gemm<<<dim3(MTOT / 128, CM / 128), blk, 0, stream>>>(
        yb, w_proj, b_proj, out);
}
